// SimpleLM_40673340293285
// MI455X (gfx1250) — matrix operand order repaired and checked by reading, not compiled, not run on hardware
//
#include <hip/hip_runtime.h>
#include <hip/hip_bf16.h>

typedef _Float16 half_t;
typedef __attribute__((ext_vector_type(16))) _Float16 v16h;
typedef __attribute__((ext_vector_type(4)))  _Float16 v4h;
typedef __attribute__((ext_vector_type(8)))  float    v8f;
typedef __attribute__((ext_vector_type(4)))  float    v4f;
typedef __attribute__((ext_vector_type(4)))  int      v4i;

// 32-byte fragment: two 16B chunks viewed as v16h for the WMMA builtins.
union AB16 {
    v16h h;
    struct { v4i lo; v4i hi; } q;
};
static_assert(sizeof(AB16) == 32, "fragment must be 32 bytes");

// ---------------------------------------------------------------------------
// Per-row symmetric 2-bit fake quantization (codes in {-2,-1,0,1}).
// One 256-thread block per row. scale = max|w| (QMAX==1), clamped >= 1e-8.
// Codes stored exactly in fp16; scale folded into the GEMM epilogue.
// Vectorized: b128 f32 loads, b64 packed-fp16 stores.
// ---------------------------------------------------------------------------
__global__ __launch_bounds__(256)
void quant_rows(const float* __restrict__ W,
                half_t* __restrict__ Wq,
                float* __restrict__ scaleOut,
                int K)
{
    __shared__ float red[256];
    const int row = blockIdx.x;
    const v4f* w4 = (const v4f*)(W + (size_t)row * K);
    const int K4 = K >> 2;

    float m = 0.0f;
    for (int k = threadIdx.x; k < K4; k += 256) {
        v4f v = w4[k];
        m = fmaxf(m, fmaxf(fmaxf(fabsf(v.x), fabsf(v.y)),
                           fmaxf(fabsf(v.z), fabsf(v.w))));
    }
    red[threadIdx.x] = m;
    __syncthreads();
#pragma unroll
    for (int s = 128; s > 0; s >>= 1) {
        if (threadIdx.x < s)
            red[threadIdx.x] = fmaxf(red[threadIdx.x], red[threadIdx.x + s]);
        __syncthreads();
    }
    const float scale = fmaxf(red[0], 1e-8f);
    if (threadIdx.x == 0) scaleOut[row] = scale;

    v4h* out4 = (v4h*)(Wq + (size_t)row * K);
    for (int k = threadIdx.x; k < K4; k += 256) {
        v4f v = w4[k];
        v4h o;
        o.x = (half_t)fminf(fmaxf(rintf(v.x / scale), -2.0f), 1.0f);
        o.y = (half_t)fminf(fmaxf(rintf(v.y / scale), -2.0f), 1.0f);
        o.z = (half_t)fminf(fmaxf(rintf(v.z / scale), -2.0f), 1.0f);
        o.w = (half_t)fminf(fmaxf(rintf(v.w / scale), -2.0f), 1.0f);
        out4[k] = o;
    }
}

// ---------------------------------------------------------------------------
// Embedding gather: h0[token][:] = (fp16) emb[x[token]][:]   (vectorized)
// ---------------------------------------------------------------------------
__global__ __launch_bounds__(256)
void embed_gather(const int* __restrict__ x,
                  const float* __restrict__ emb,
                  half_t* __restrict__ h0,
                  int E)
{
    const int tok = blockIdx.x;
    const int v   = x[tok];
    const v4f* src = (const v4f*)(emb + (size_t)v * E);
    v4h* dst = (v4h*)(h0 + (size_t)tok * E);
    const int E4 = E >> 2;
    for (int k = threadIdx.x; k < E4; k += 256) {
        v4f f = src[k];
        v4h h;
        h.x = (half_t)f.x;
        h.y = (half_t)f.y;
        h.z = (half_t)f.z;
        h.w = (half_t)f.w;
        dst[k] = h;
    }
}

// ---------------------------------------------------------------------------
// Quantized GEMM: out[m][n] = epilogue( sum_k A[m][k] * Bw[n][k], n )
//   A  : [M x K] fp16 activations (row-major)
//   Bw : [N x K] fp16 quantization codes (row-major)
// With both operands row-major along K, the WMMA A and B fragments are
// contiguous 16B chunks per lane -> everything loads straight from global
// with b128s; no LDS staging, no barriers, fully pipelineable K loop.
// Workgroup: 256 threads = 8 waves -> 64(M) x 128(N) tile.
// Wave (4M x 2N arrangement): 16(M) x 64(N) -> 4 accumulators v8f.
// Epilogue: v = scale[n]*acc + bias[n]; ReLU+fp16 store or NT f32 store.
// ---------------------------------------------------------------------------
template<bool RELU_H16>
__global__ __launch_bounds__(256)
void gemm_q2(const half_t* __restrict__ A,
             const half_t* __restrict__ Bw,
             const float*  __restrict__ scl,
             const float*  __restrict__ bias,
             float*  __restrict__ outF,
             half_t* __restrict__ outH,
             int M, int N, int K)
{
    const int tid   = threadIdx.x;
    const int lane  = tid & 31;
    const int wave  = tid >> 5;
    const int waveM = wave & 3;     // 0..3
    const int waveN = wave >> 2;    // 0..1

    const int mBlock = blockIdx.x * 64;
    const int nBlock = blockIdx.y * 128;

    // Fragment geometry (wave32; contiguous K per lane half is this code's order, same for A and B):
    const int mLane  = lane & 15;       // A: M = lane&15 ; B/C/D: N = lane&15
    const int hi     = lane >> 4;       // lane-group selector
    const int koffA  = hi * 16;         // A halves: K[koff .. koff+15]
    const int kbaseB = hi * 16;         // B halves: K[kbase .. kbase+15]

    // Per-lane A row pointer for this wave's 16-row strip.
    const half_t* ap = A + (size_t)(mBlock + waveM * 16 + mLane) * K + koffA;

    // Per-lane B row pointers for this wave's 4 N-tiles.
    const int nLane = nBlock + waveN * 64 + mLane;
    const half_t* bp0 = Bw + (size_t)(nLane +  0) * K + kbaseB;
    const half_t* bp1 = Bw + (size_t)(nLane + 16) * K + kbaseB;
    const half_t* bp2 = Bw + (size_t)(nLane + 32) * K + kbaseB;
    const half_t* bp3 = Bw + (size_t)(nLane + 48) * K + kbaseB;

    v8f acc[4] = {};

    for (int k0 = 0; k0 < K; k0 += 32) {
        // Prefetch ~4 iterations ahead (speculative global_prefetch_b8).
        __builtin_prefetch(bp0 + k0 + 128, 0, 1);
        __builtin_prefetch(bp1 + k0 + 128, 0, 1);
        __builtin_prefetch(bp2 + k0 + 128, 0, 1);
        __builtin_prefetch(bp3 + k0 + 128, 0, 1);

        AB16 af;
        af.q.lo = *(const v4i*)(ap + k0);
        af.q.hi = *(const v4i*)(ap + k0 + 8);

        AB16 bf0, bf1, bf2, bf3;
        bf0.q.lo = *(const v4i*)(bp0 + k0);
        bf0.q.hi = *(const v4i*)(bp0 + k0 + 8);
        bf1.q.lo = *(const v4i*)(bp1 + k0);
        bf1.q.hi = *(const v4i*)(bp1 + k0 + 8);
        bf2.q.lo = *(const v4i*)(bp2 + k0);
        bf2.q.hi = *(const v4i*)(bp2 + k0 + 8);
        bf3.q.lo = *(const v4i*)(bp3 + k0);
        bf3.q.hi = *(const v4i*)(bp3 + k0 + 8);

        acc[0] = __builtin_amdgcn_wmma_f32_16x16x32_f16(false, af.h, false, bf0.h,
                                                        (short)0, acc[0], false, false);
        acc[1] = __builtin_amdgcn_wmma_f32_16x16x32_f16(false, af.h, false, bf1.h,
                                                        (short)0, acc[1], false, false);
        acc[2] = __builtin_amdgcn_wmma_f32_16x16x32_f16(false, af.h, false, bf2.h,
                                                        (short)0, acc[2], false, false);
        acc[3] = __builtin_amdgcn_wmma_f32_16x16x32_f16(false, af.h, false, bf3.h,
                                                        (short)0, acc[3], false, false);
    }

    // Epilogue. C/D layout: VGPR j -> row (j + 8*hi), col = lane&15.
    const int rowBase = mBlock + waveM * 16 + hi * 8;
    const int colBase = nBlock + waveN * 64 + mLane;
#pragma unroll
    for (int t = 0; t < 4; ++t) {
        const int n  = colBase + t * 16;
        const float s  = scl[n];
        const float bb = bias[n];
#pragma unroll
        for (int j = 0; j < 8; ++j) {
            float v = fmaf(s, acc[t][j], bb);
            const size_t o = (size_t)(rowBase + j) * (size_t)N + (size_t)n;
            if constexpr (RELU_H16) {
                v = v > 0.0f ? v : 0.0f;
                outH[o] = (half_t)v;        // h1 is re-read: keep temporal
            } else {
                // Logits are write-once: non-temporal, keep L2 for A/B reuse.
                __builtin_nontemporal_store(v, &outF[o]);
            }
        }
    }
}

// ---------------------------------------------------------------------------
// Host-side launcher.
// Inputs (setup_inputs order): x[int32 4096], emb[f32 32000x1024],
//   W1[f32 4096x1024], b1[f32 4096], W2[f32 32000x4096], b2[f32 32000]
// Output: f32 [4,1024,32000]
// ---------------------------------------------------------------------------
extern "C" void kernel_launch(void* const* d_in, const int* in_sizes, int n_in,
                              void* d_out, int out_size, void* d_ws, size_t ws_size,
                              hipStream_t stream)
{
    (void)in_sizes; (void)n_in; (void)out_size; (void)ws_size;

    const int* x     = (const int*)d_in[0];
    const float* emb = (const float*)d_in[1];
    const float* W1  = (const float*)d_in[2];
    const float* b1  = (const float*)d_in[3];
    const float* W2  = (const float*)d_in[4];
    const float* b2  = (const float*)d_in[5];
    float* out = (float*)d_out;

    const size_t Vv = 32000, Ee = 1024, Hh = 4096, Mm = 4096; // Mm = B*S

    // Workspace carve-out (deterministic layout, 256B aligned).
    size_t off = 0;
    char* ws = (char*)d_ws;
    auto take = [&](size_t bytes) -> void* {
        void* p = ws + off;
        off += (bytes + 255) & ~(size_t)255;
        return p;
    };
    half_t* Wq1 = (half_t*)take(Hh * Ee * sizeof(half_t));   //   8 MB
    float*  s1  = (float* )take(Hh * sizeof(float));
    half_t* Wq2 = (half_t*)take(Vv * Hh * sizeof(half_t));   // 262 MB
    float*  s2  = (float* )take(Vv * sizeof(float));
    half_t* h0  = (half_t*)take(Mm * Ee * sizeof(half_t));   //   8 MB
    half_t* h1  = (half_t*)take(Mm * Hh * sizeof(half_t));   //  33 MB

    // 1) Quantize both weight matrices (per-row 2-bit codes in fp16 + scales).
    quant_rows<<<dim3((unsigned)Hh), 256, 0, stream>>>(W1, Wq1, s1, (int)Ee);
    quant_rows<<<dim3((unsigned)Vv), 256, 0, stream>>>(W2, Wq2, s2, (int)Hh);

    // 2) Embedding gather to fp16 activations.
    embed_gather<<<dim3((unsigned)Mm), 256, 0, stream>>>(x, emb, h0, (int)Ee);

    // 3) h1 = relu(h0 @ Wq1^T * s1 + b1)   [4096 x 4096] fp16
    gemm_q2<true><<<dim3((unsigned)(Mm / 64), (unsigned)(Hh / 128)), 256, 0, stream>>>(
        h0, Wq1, s1, b1, nullptr, h1, (int)Mm, (int)Hh, (int)Ee);

    // 4) logits = h1 @ Wq2^T * s2 + b2     [4096 x 32000] f32
    gemm_q2<false><<<dim3((unsigned)(Mm / 64), (unsigned)(Vv / 128)), 256, 0, stream>>>(
        h1, Wq2, s2, b2, out, nullptr, (int)Mm, (int)Vv, (int)Hh);
}
